// LSTMDecoder_11716670784118
// MI455X (gfx1250) — matrix operand order repaired and checked by reading, not compiled, not run on hardware
//
#include <hip/hip_runtime.h>
#include <hip/hip_bf16.h>

// ---------------------------------------------------------------------------
// CDNA5 (gfx1250) attention-LSTM decoder.
// Per-step GEMMs (ctxE, mix, gates, logits) on v_wmma_f32_16x16x32_bf16 with
// f32 accumulation. One wave per 16-wide N-tile computing BOTH 16-row M-tiles
// (M=32 batch) -> each B fragment is loaded once and reused by two WMMAs,
// halving the dominant L2 weight traffic. k_logits additionally stages the
// 32KB activation matrix into LDS with global_load_async_to_lds_b128
// (ASYNCcnt path) so A-fragments come from LDS while B streams from L2.
// ---------------------------------------------------------------------------

typedef __bf16 bf16_t;
typedef __attribute__((ext_vector_type(16))) __bf16 v16bf;
typedef __attribute__((ext_vector_type(8)))  __bf16 v8bf;
typedef __attribute__((ext_vector_type(8)))  float  v8f;

#define B_  32
#define S_  49
#define T_  64
#define E_  512
#define H_  512
#define DI_ 2048
#define A_  128
#define V_  32000
#define LN_EPS 1e-5f

// ---------------- WMMA fragment helpers (A/B: contiguous K per lane half, same order in both) ----------------
// A: 16x32 bf16, lane L holds row M=L&15; K range (L>=16)*16 .. +15.
__device__ __forceinline__ v16bf load_a_frag(const bf16_t* A, int lda,
                                             int mtile, int k0, int lane) {
  int m  = mtile * 16 + (lane & 15);
  int kb = (lane >> 4) * 16;
  const bf16_t* p = A + (size_t)m * lda + k0 + kb;
  v8bf lo = *(const v8bf*)(p);        // K = k0+kb .. +7
  v8bf hi = *(const v8bf*)(p + 8);    // K = k0+kb+8 .. +7
  v16bf a;
#pragma unroll
  for (int i = 0; i < 8; ++i) { a[i] = lo[i]; a[8 + i] = hi[i]; }
  return a;
}

// B: 32x16 bf16 (W stored row-major [N][K], B = W^T). Lane L holds col N=L&15,
// K range (L>=16)*16 .. +15 -> one contiguous 32B load along K.
__device__ __forceinline__ v16bf load_b_frag(const bf16_t* __restrict__ W, int ldw,
                                             int ntile, int k0, int lane) {
  int n  = ntile * 16 + (lane & 15);
  int kb = (lane >> 4) * 16;
  return *(const v16bf*)(W + (size_t)n * ldw + k0 + kb);
}

// Dual-M-tile accumulate: load B once, issue two WMMAs (M-tiles 0 and 1).
__device__ __forceinline__ void wmma_acc2(const bf16_t* Act, int lda,
                                          const bf16_t* __restrict__ W, int ldw, int K,
                                          int ntile, int lane, v8f& c0, v8f& c1) {
  int n = ntile * 16 + (lane & 15);
  for (int k0 = 0; k0 < K; k0 += 32) {
    if (k0 + 32 < K)  // pull next B K-slab toward the WGP (global_prefetch_b8)
      __builtin_prefetch(W + (size_t)n * ldw + k0 + 32, 0, 3);
    v16bf b  = load_b_frag(W, ldw, ntile, k0, lane);
    v16bf a0 = load_a_frag(Act, lda, 0, k0, lane);
    v16bf a1 = load_a_frag(Act, lda, 1, k0, lane);
    c0 = __builtin_amdgcn_wmma_f32_16x16x32_bf16(false, a0, false, b, (short)0, c0,
                                                 false, false);
    c1 = __builtin_amdgcn_wmma_f32_16x16x32_bf16(false, a1, false, b, (short)0, c1,
                                                 false, false);
  }
}

// C/D layout: lane L -> col n=L&15, rows m = (L>=16?8:0)+r  (r = vgpr index)
__device__ __forceinline__ int cd_col(int ntile, int lane) { return ntile * 16 + (lane & 15); }
__device__ __forceinline__ int cd_row0(int mtile, int lane) { return mtile * 16 + ((lane >> 4) << 3); }

// ---------------------------- precompute kernels ----------------------------
__global__ void k_convert(const float* __restrict__ src, bf16_t* __restrict__ dst, int n) {
  int i = blockIdx.x * blockDim.x + threadIdx.x;
  if (i < n) dst[i] = (bf16_t)src[i];
}

// proj_image[b,s,a] = sum_d img[b,s,d] * W_img[a,d]   (one-time)
__global__ void k_proj(const float* __restrict__ img, const float* __restrict__ W_img,
                       float* __restrict__ proj) {
  int idx = blockIdx.x * blockDim.x + threadIdx.x;       // B_*S_*A_ = 200704
  if (idx >= B_ * S_ * A_) return;
  int a  = idx & (A_ - 1);
  int bs = idx >> 7;
  const float* x = img  + (size_t)bs * DI_;
  const float* w = W_img + (size_t)a * DI_;
  float acc = 0.f;
  for (int d = 0; d < DI_; ++d) acc += x[d] * w[d];
  proj[idx] = acc;
}

// h0 = tanh(mean_s(img) @ W_h0.T + b_h0)*0.5 ; c0 likewise   (one-time)
__global__ void k_init(const float* __restrict__ img,
                       const float* __restrict__ W_h0, const float* __restrict__ b_h0,
                       const float* __restrict__ W_c0, const float* __restrict__ b_c0,
                       float* __restrict__ h, float* __restrict__ c,
                       bf16_t* __restrict__ hbf) {
  int b = blockIdx.x, tid = threadIdx.x;                 // 32 blocks x 256
  __shared__ float g[DI_];
  for (int d = tid; d < DI_; d += 256) {
    float acc = 0.f;
    for (int s = 0; s < S_; ++s) acc += img[((size_t)b * S_ + s) * DI_ + d];
    g[d] = acc * (1.f / (float)S_);
  }
  __syncthreads();
  for (int j = tid; j < H_; j += 256) {
    float ah = b_h0[j], ac = b_c0[j];
    const float* wh = W_h0 + (size_t)j * DI_;
    const float* wc = W_c0 + (size_t)j * DI_;
    for (int d = 0; d < DI_; ++d) { float gv = g[d]; ah += gv * wh[d]; ac += gv * wc[d]; }
    float hv = tanhf(ah) * 0.5f, cv = tanhf(ac) * 0.5f;
    h[b * H_ + j] = hv; c[b * H_ + j] = cv; hbf[b * H_ + j] = (bf16_t)hv;
  }
}

// ----------------------------- per-step kernels -----------------------------
// Additive attention: ph = h@W_hid.T ; w = softmax(tanh(proj+ph)@w_score) ;
// ctx = w . img_seq  -> ctx_bf16 [B, DI]
__global__ void k_attn(const float* __restrict__ h, const float* __restrict__ W_hid,
                       const float* __restrict__ proj, const float* __restrict__ w_score,
                       const float* __restrict__ img, bf16_t* __restrict__ ctxbf) {
  int b = blockIdx.x, tid = threadIdx.x;                 // 32 blocks x 256
  __shared__ float hrow[H_];
  __shared__ float ph[A_];
  __shared__ float sc[S_];
  __shared__ float wts[S_];
  hrow[tid] = h[b * H_ + tid];
  hrow[256 + tid] = h[b * H_ + 256 + tid];
  __syncthreads();
  if (tid < A_) {
    const float* w = W_hid + (size_t)tid * H_;
    float acc = 0.f;
    for (int k = 0; k < H_; ++k) acc += hrow[k] * w[k];
    ph[tid] = acc;
  }
  __syncthreads();
  if (tid < S_) {
    const float* p = proj + ((size_t)b * S_ + tid) * A_;
    float acc = 0.f;
    for (int a = 0; a < A_; ++a) acc += tanhf(p[a] + ph[a]) * w_score[a];
    sc[tid] = acc;
  }
  __syncthreads();
  if (tid < S_) {
    float mx = -1e30f;
    for (int s = 0; s < S_; ++s) mx = fmaxf(mx, sc[s]);
    float sum = 0.f;
    for (int s = 0; s < S_; ++s) sum += __expf(sc[s] - mx);
    wts[tid] = __expf(sc[tid] - mx) / sum;
  }
  __syncthreads();
  for (int d = tid; d < DI_; d += 256) {
    float acc = 0.f;
    for (int s = 0; s < S_; ++s) acc += wts[s] * img[((size_t)b * S_ + s) * DI_ + d];
    ctxbf[b * DI_ + d] = (bf16_t)acc;
  }
}

// embedding lookup (+PAD mask): emb fp32 (residual) + left half of xcat (bf16)
__global__ void k_embed(const int* __restrict__ toks, const float* __restrict__ embed,
                        float* __restrict__ emb, bf16_t* __restrict__ xcat, int t) {
  int idx = blockIdx.x * blockDim.x + threadIdx.x;       // B_*E_ = 16384
  if (idx >= B_ * E_) return;
  int b = idx >> 9, e = idx & (E_ - 1);
  int tok = toks[b * T_ + t];
  float v = (tok == 0) ? 0.f : embed[(size_t)tok * E_ + e];
  emb[idx] = v;
  xcat[b * (2 * E_) + e] = (bf16_t)v;
}

// ctxE = ctx @ W_ctx.T -> right half of xcat (bf16).  M=32,N=512,K=2048
__global__ void k_ctxE(const bf16_t* __restrict__ ctx, const bf16_t* __restrict__ Wctx,
                       bf16_t* __restrict__ xcat) {
  int ntile = (blockIdx.x * blockDim.x + threadIdx.x) >> 5;   // 32 ntiles
  int lane  = threadIdx.x & 31;
  v8f c0 = {}, c1 = {};
  wmma_acc2(ctx, DI_, Wctx, DI_, DI_, ntile, lane, c0, c1);
  int n = cd_col(ntile, lane);
  int r0 = cd_row0(0, lane), r1 = cd_row0(1, lane);
#pragma unroll
  for (int r = 0; r < 8; ++r) {
    xcat[(r0 + r) * (2 * E_) + E_ + n] = (bf16_t)c0[r];
    xcat[(r1 + r) * (2 * E_) + E_ + n] = (bf16_t)c1[r];
  }
}

// x = xcat @ W_mix.T (pre-activation, fp32).  M=32,N=512,K=1024
__global__ void k_mix(const bf16_t* __restrict__ xcat, const bf16_t* __restrict__ Wmix,
                      float* __restrict__ xmix) {
  int ntile = (blockIdx.x * blockDim.x + threadIdx.x) >> 5;   // 32 ntiles
  int lane  = threadIdx.x & 31;
  v8f c0 = {}, c1 = {};
  wmma_acc2(xcat, 2 * E_, Wmix, 2 * E_, 2 * E_, ntile, lane, c0, c1);
  int n = cd_col(ntile, lane);
  int r0 = cd_row0(0, lane), r1 = cd_row0(1, lane);
#pragma unroll
  for (int r = 0; r < 8; ++r) {
    xmix[(r0 + r) * E_ + n] = c0[r];
    xmix[(r1 + r) * E_ + n] = c1[r];
  }
}

// relu -> LayerNorm -> *g+b -> +emb -> bf16
__global__ void k_ln(const float* __restrict__ xmix, const float* __restrict__ ln_g,
                     const float* __restrict__ ln_b, const float* __restrict__ emb,
                     bf16_t* __restrict__ xln) {
  int b = blockIdx.x, tid = threadIdx.x;                 // 32 blocks x 256
  __shared__ float red[256];
  float v0 = fmaxf(xmix[b * E_ + tid], 0.f);
  float v1 = fmaxf(xmix[b * E_ + 256 + tid], 0.f);
  red[tid] = v0 + v1;
  __syncthreads();
  for (int o = 128; o > 0; o >>= 1) { if (tid < o) red[tid] += red[tid + o]; __syncthreads(); }
  float mu = red[0] * (1.f / (float)E_);
  __syncthreads();
  float d0 = v0 - mu, d1 = v1 - mu;
  red[tid] = d0 * d0 + d1 * d1;
  __syncthreads();
  for (int o = 128; o > 0; o >>= 1) { if (tid < o) red[tid] += red[tid + o]; __syncthreads(); }
  float rstd = rsqrtf(red[0] * (1.f / (float)E_) + LN_EPS);
  float y0 = d0 * rstd * ln_g[tid]       + ln_b[tid]       + emb[b * E_ + tid];
  float y1 = d1 * rstd * ln_g[256 + tid] + ln_b[256 + tid] + emb[b * E_ + 256 + tid];
  xln[b * E_ + tid]       = (bf16_t)y0;
  xln[b * E_ + 256 + tid] = (bf16_t)y1;
}

// gates = xln@W_ih.T + h@W_hh.T + b_ih + b_hh.  M=32,N=2048, two K=512 passes
__global__ void k_gates(const bf16_t* __restrict__ xln, const bf16_t* __restrict__ hbf,
                        const bf16_t* __restrict__ Wih, const bf16_t* __restrict__ Whh,
                        const float* __restrict__ b_ih, const float* __restrict__ b_hh,
                        float* __restrict__ gates) {
  int ntile = (blockIdx.x * blockDim.x + threadIdx.x) >> 5;   // 128 ntiles
  int lane  = threadIdx.x & 31;
  v8f c0 = {}, c1 = {};
  wmma_acc2(xln, E_, Wih, E_, E_, ntile, lane, c0, c1);
  wmma_acc2(hbf, H_, Whh, H_, H_, ntile, lane, c0, c1);
  int n = cd_col(ntile, lane);
  int r0 = cd_row0(0, lane), r1 = cd_row0(1, lane);
  float bias = b_ih[n] + b_hh[n];
#pragma unroll
  for (int r = 0; r < 8; ++r) {
    gates[(r0 + r) * (4 * H_) + n] = c0[r] + bias;
    gates[(r1 + r) * (4 * H_) + n] = c1[r] + bias;
  }
}

// LSTM cell (torch gate order i,f,g,o)
__global__ void k_cell(const float* __restrict__ gates, float* __restrict__ h,
                       float* __restrict__ c, bf16_t* __restrict__ hbf) {
  int idx = blockIdx.x * blockDim.x + threadIdx.x;       // B_*H_ = 16384
  if (idx >= B_ * H_) return;
  int b = idx >> 9, j = idx & (H_ - 1);
  const float* g = gates + (size_t)b * 4 * H_;
  float ig = g[j], fg = g[H_ + j], gg = g[2 * H_ + j], og = g[3 * H_ + j];
  float si = 1.f / (1.f + __expf(-ig));
  float sf = 1.f / (1.f + __expf(-fg));
  float so = 1.f / (1.f + __expf(-og));
  float c2 = sf * c[idx] + si * tanhf(gg);
  float h2 = so * tanhf(c2);
  c[idx] = c2;
  h[idx] = h2;
  hbf[idx] = (bf16_t)h2;
}

// logits = h2 @ W_head.T -> out[b,t,:].  M=32,N=32000,K=512 (the big one).
// Stage the 32KB A matrix (h2, bf16) into LDS once per block with
// global_load_async_to_lds_b128 (ASYNCcnt), then A-frags come from LDS.
__global__ void k_logits(const bf16_t* __restrict__ hbf, const bf16_t* __restrict__ Wh,
                         float* __restrict__ out, int t) {
  __shared__ bf16_t smA[B_ * H_];                        // 32 KB
  int tid  = threadIdx.x;                                // 128 threads, 4 waves
  // async-copy A: 32768 B / (128 thr * 16 B) = 16 b128 transfers per thread
#pragma unroll
  for (int byteOff = tid * 16; byteOff < B_ * H_ * 2; byteOff += 128 * 16) {
    const bf16_t* gsrc = hbf + (byteOff >> 1);
    unsigned ldsAddr = (unsigned)(size_t)(&smA[0]) + (unsigned)byteOff;
    asm volatile("global_load_async_to_lds_b128 %0, %1, off"
                 :: "v"(ldsAddr), "v"(gsrc) : "memory");
  }
  asm volatile("s_wait_asynccnt 0x0" ::: "memory");
  __syncthreads();

  int ntile = (blockIdx.x * blockDim.x + tid) >> 5;      // 2000 ntiles
  int lane  = tid & 31;
  v8f c0 = {}, c1 = {};
  wmma_acc2(smA, H_, Wh, H_, H_, ntile, lane, c0, c1);   // A from LDS (ds path)
  int n = cd_col(ntile, lane);
  int r0 = cd_row0(0, lane), r1 = cd_row0(1, lane);
#pragma unroll
  for (int r = 0; r < 8; ++r) {
    out[(size_t)(r0 + r) * (T_ * V_) + (size_t)t * V_ + n] = c0[r];
    out[(size_t)(r1 + r) * (T_ * V_) + (size_t)t * V_ + n] = c1[r];
  }
}

// --------------------------------- launch -----------------------------------
extern "C" void kernel_launch(void* const* d_in, const int* in_sizes, int n_in,
                              void* d_out, int out_size, void* d_ws, size_t ws_size,
                              hipStream_t stream) {
  const float* img     = (const float*)d_in[0];
  const int*   toks    = (const int*)  d_in[1];
  const float* embed   = (const float*)d_in[2];
  const float* W_ctx   = (const float*)d_in[3];
  const float* W_mix   = (const float*)d_in[4];
  const float* ln_g    = (const float*)d_in[5];
  const float* ln_b    = (const float*)d_in[6];
  const float* W_ih    = (const float*)d_in[7];
  const float* W_hh    = (const float*)d_in[8];
  const float* b_ih    = (const float*)d_in[9];
  const float* b_hh    = (const float*)d_in[10];
  const float* W_head  = (const float*)d_in[11];
  const float* W_img   = (const float*)d_in[12];
  const float* W_hid   = (const float*)d_in[13];
  const float* w_score = (const float*)d_in[14];
  const float* W_h0    = (const float*)d_in[15];
  const float* b_h0    = (const float*)d_in[16];
  const float* W_c0    = (const float*)d_in[17];
  const float* b_c0    = (const float*)d_in[18];
  float* out = (float*)d_out;

  // workspace carve-out (256B aligned)
  char* ws = (char*)d_ws;
  auto alloc = [&](size_t bytes) -> void* {
    void* p = (void*)ws;
    ws += (bytes + 255) & ~(size_t)255;
    return p;
  };
  bf16_t* Whb   = (bf16_t*)alloc((size_t)V_ * H_ * 2);        // 32.8 MB
  bf16_t* Wihb  = (bf16_t*)alloc((size_t)4 * H_ * E_ * 2);
  bf16_t* Whhb  = (bf16_t*)alloc((size_t)4 * H_ * H_ * 2);
  bf16_t* Wmixb = (bf16_t*)alloc((size_t)E_ * 2 * E_ * 2);
  bf16_t* Wctxb = (bf16_t*)alloc((size_t)E_ * DI_ * 2);
  float*  proj  = (float*) alloc((size_t)B_ * S_ * A_ * 4);
  float*  hf    = (float*) alloc((size_t)B_ * H_ * 4);
  float*  cf    = (float*) alloc((size_t)B_ * H_ * 4);
  bf16_t* hbf   = (bf16_t*)alloc((size_t)B_ * H_ * 2);
  bf16_t* ctxbf = (bf16_t*)alloc((size_t)B_ * DI_ * 2);
  bf16_t* xcat  = (bf16_t*)alloc((size_t)B_ * 2 * E_ * 2);
  float*  embf  = (float*) alloc((size_t)B_ * E_ * 4);
  float*  xmix  = (float*) alloc((size_t)B_ * E_ * 4);
  bf16_t* xln   = (bf16_t*)alloc((size_t)B_ * E_ * 2);
  float*  gates = (float*) alloc((size_t)B_ * 4 * H_ * 4);

  auto cvt = [&](const float* s, bf16_t* d, int n) {
    k_convert<<<(n + 255) / 256, 256, 0, stream>>>(s, d, n);
  };

  // ---- one-time precompute ----
  cvt(W_head, Whb,   V_ * H_);
  cvt(W_ih,   Wihb,  4 * H_ * E_);
  cvt(W_hh,   Whhb,  4 * H_ * H_);
  cvt(W_mix,  Wmixb, E_ * 2 * E_);
  cvt(W_ctx,  Wctxb, E_ * DI_);
  k_proj<<<(B_ * S_ * A_ + 255) / 256, 256, 0, stream>>>(img, W_img, proj);
  k_init<<<B_, 256, 0, stream>>>(img, W_h0, b_h0, W_c0, b_c0, hf, cf, hbf);

  // ---- 64 sequential decode steps (stream order = dependency order) ----
  for (int t = 0; t < T_; ++t) {
    k_attn  <<<B_, 256, 0, stream>>>(hf, W_hid, proj, w_score, img, ctxbf);
    k_embed <<<(B_ * E_ + 255) / 256, 256, 0, stream>>>(toks, embed, embf, xcat, t);
    k_ctxE  <<<8,   128, 0, stream>>>(ctxbf, Wctxb, xcat);    // 32 waves
    k_mix   <<<8,   128, 0, stream>>>(xcat, Wmixb, xmix);     // 32 waves
    k_ln    <<<B_,  256, 0, stream>>>(xmix, ln_g, ln_b, embf, xln);
    k_gates <<<32,  128, 0, stream>>>(xln, hbf, Wihb, Whhb, b_ih, b_hh, gates); // 128 waves
    k_cell  <<<(B_ * H_ + 255) / 256, 256, 0, stream>>>(gates, hf, cf, hbf);
    k_logits<<<500, 128, 0, stream>>>(hbf, Whb, out, t);      // 2000 waves
  }
}
